// BitLinearOptimized_79293686219430
// MI455X (gfx1250) — compile-verified, matrix operand order checked by reading, not run on hardware
//
#include <hip/hip_runtime.h>

typedef __attribute__((ext_vector_type(16))) _Float16 v16h;
typedef __attribute__((ext_vector_type(8)))  _Float16 v8h;
typedef __attribute__((ext_vector_type(8)))  float    v8f;
typedef __attribute__((ext_vector_type(4)))  float    v4f;
typedef int v4i __attribute__((vector_size(16)));   // exact builtin param type

#define N_ROWS 8192
#define IN_F   4096
#define OUT_F  4096
#define GRP    1024   /* IN_F / 4 */

#define BLK_M  128
#define BLK_N  256
#define BLK_K  32
#define LDS_STRIDE 40   /* 32 + 8 halves: 80B rows -> 16B aligned b128, conflict-free */

#if __has_builtin(__builtin_amdgcn_global_load_async_to_lds_b128) && \
    __has_builtin(__builtin_amdgcn_s_wait_asynccnt)
#define ASYNC_LDS 1
#else
#define ASYNC_LDS 0
#endif

#define LDS_AS __attribute__((address_space(3)))

// ---------------------------------------------------------------------------
// Kernel 0: zero the absmax scratch word (ws is poisoned to 0xAA by harness)
// ---------------------------------------------------------------------------
__global__ void init_ws_kernel(unsigned int* absmax_bits) {
    if (threadIdx.x == 0) absmax_bits[0] = 0u;
}

// ---------------------------------------------------------------------------
// Kernel 1: global absmax over input (bit-pattern atomicMax is order-
// preserving for non-negative floats). float4 loads, wave32 shuffle tree.
// Regular-temporal loads on purpose: input is re-read by the quantize pass
// and (128MB) fits the 192MB L2.
// ---------------------------------------------------------------------------
__global__ __launch_bounds__(256)
void absmax_kernel(const float* __restrict__ x, unsigned int* __restrict__ out, int n4) {
    int i      = blockIdx.x * blockDim.x + threadIdx.x;
    int stride = gridDim.x * blockDim.x;
    const v4f* x4 = reinterpret_cast<const v4f*>(x);
    float m = 0.0f;
    for (int idx = i; idx < n4; idx += stride) {
        v4f v = x4[idx];
        m = fmaxf(m, fmaxf(fmaxf(fabsf(v.x), fabsf(v.y)),
                           fmaxf(fabsf(v.z), fabsf(v.w))));
    }
    #pragma unroll
    for (int off = 16; off > 0; off >>= 1)
        m = fmaxf(m, __shfl_xor(m, off, 32));
    if ((threadIdx.x & 31) == 0)
        atomicMax(out, __float_as_uint(m));
}

// ---------------------------------------------------------------------------
// Kernel 2: quantize (RNE via rintf, clip [-128,127]) + group-of-4 mean.
// One thread per group -> one float4 load, one f16 store. Exact in f16.
// ---------------------------------------------------------------------------
__global__ __launch_bounds__(256)
void quant_group_kernel(const float* __restrict__ x,
                        const unsigned int* __restrict__ absmax_bits,
                        _Float16* __restrict__ Xm, int total_groups) {
    int i = blockIdx.x * blockDim.x + threadIdx.x;
    if (i >= total_groups) return;
    float amax  = __uint_as_float(absmax_bits[0]);
    float inv_s = 127.0f / amax;
    v4f v = reinterpret_cast<const v4f*>(x)[i];
    float q0 = fminf(fmaxf(rintf(v.x * inv_s), -128.0f), 127.0f);
    float q1 = fminf(fmaxf(rintf(v.y * inv_s), -128.0f), 127.0f);
    float q2 = fminf(fmaxf(rintf(v.z * inv_s), -128.0f), 127.0f);
    float q3 = fminf(fmaxf(rintf(v.w * inv_s), -128.0f), 127.0f);
    Xm[i] = (_Float16)((q0 + q1 + q2 + q3) * 0.25f);
}

// ---------------------------------------------------------------------------
// Kernel 3: ternary weight group-of-4 sum -> f16 (values in [-4,4], exact).
// Weights (64MB) are read exactly once: non-temporal loads keep them from
// evicting the L2-resident input / Xm / Ws working set.
// ---------------------------------------------------------------------------
__global__ __launch_bounds__(256)
void wsum_group_kernel(const float* __restrict__ w,
                       _Float16* __restrict__ Ws, int total_groups) {
    int i = blockIdx.x * blockDim.x + threadIdx.x;
    if (i >= total_groups) return;
    v4f v = __builtin_nontemporal_load(reinterpret_cast<const v4f*>(w) + i);
    Ws[i] = (_Float16)(v.x + v.y + v.z + v.w);
}

// ---------------------------------------------------------------------------
// Kernel 4: WMMA GEMM  out[N_ROWS,OUT_F] = Xm[N_ROWS,GRP] * Ws[OUT_F,GRP]^T
// 128x256 block tile, 8 waves (2Mx4N) of 64x64 wave tiles (4x4 subtiles),
// K-step 32, double-buffered LDS (60KB), software pipelined via
// GLOBAL_LOAD_ASYNC_TO_LDS_B128 (ASYNCcnt). The pipeline loop is kept
// NOT-unrolled so fragment live ranges stay within 256 VGPRs (avoids the
// s_set_vgpr_msb + v_mov shuffle storm seen with cross-iteration pipelining).
// Output stores are non-temporal (128MB streamed once, never re-read).
// ---------------------------------------------------------------------------
__global__ __launch_bounds__(256)
void bitlinear_wmma_gemm(const _Float16* __restrict__ Xm,
                         const _Float16* __restrict__ Ws,
                         const float* __restrict__ bias,
                         const float* __restrict__ wscale,
                         const unsigned int* __restrict__ absmax_bits,
                         float* __restrict__ out) {
    __shared__ _Float16 As[2][BLK_M * LDS_STRIDE];   // 2 x 10,240 B
    __shared__ _Float16 Bs[2][BLK_N * LDS_STRIDE];   // 2 x 20,480 B

    const int t    = threadIdx.x;
    const int lane = t & 31;
    const int wave = t >> 5;                 // 0..7
    const int wm   = wave & 1;               // 0..1 : 64-row strip of 128
    const int wn   = wave >> 1;              // 0..3 : 64-col strip of 256

    const int m_blk = blockIdx.x * BLK_M;
    const int n_blk = blockIdx.y * BLK_N;

    const int lrow  = lane & 15;             // M (A) / N (B) within 16x16 tile
    const int khalf = (lane < 16) ? 0 : 8;   // 16-bit A/B lane K-offset

    // cooperative-load assignment: A -> 2 threads/row (16 halves each),
    //                              B -> 1 thread/row  (32 halves)
    const int a_row = t >> 1;                // 0..127
    const int a_k   = (t & 1) * 16;          // 0 | 16

    v8f acc[4][4];
    #pragma unroll
    for (int mi = 0; mi < 4; ++mi)
        #pragma unroll
        for (int ni = 0; ni < 4; ++ni)
            acc[mi][ni] = {};

#if ASYNC_LDS
    // ---- direct async global -> LDS staging (no VGPR round-trip) ----
    auto stage = [&](int buf, int kk) {
        const _Float16* pa = Xm + (size_t)(m_blk + a_row) * GRP + kk + a_k;
        _Float16* la = &As[buf][a_row * LDS_STRIDE + a_k];
        __builtin_amdgcn_global_load_async_to_lds_b128(
            (v4i*)pa,       (LDS_AS v4i*)la,       0, 0);
        __builtin_amdgcn_global_load_async_to_lds_b128(
            (v4i*)(pa + 8), (LDS_AS v4i*)(la + 8), 0, 0);
        const _Float16* pb = Ws + (size_t)(n_blk + t) * GRP + kk;
        _Float16* lb = &Bs[buf][t * LDS_STRIDE];
        __builtin_amdgcn_global_load_async_to_lds_b128(
            (v4i*)pb,        (LDS_AS v4i*)lb,        0, 0);
        __builtin_amdgcn_global_load_async_to_lds_b128(
            (v4i*)(pb + 8),  (LDS_AS v4i*)(lb + 8),  0, 0);
        __builtin_amdgcn_global_load_async_to_lds_b128(
            (v4i*)(pb + 16), (LDS_AS v4i*)(lb + 16), 0, 0);
        __builtin_amdgcn_global_load_async_to_lds_b128(
            (v4i*)(pb + 24), (LDS_AS v4i*)(lb + 24), 0, 0);
    };
#else
    v8h ar0, ar1, br0, br1, br2, br3;        // global->LDS staging registers
    auto gload = [&](int kk) {
        const _Float16* pa = Xm + (size_t)(m_blk + a_row) * GRP + kk + a_k;
        ar0 = *reinterpret_cast<const v8h*>(pa);
        ar1 = *reinterpret_cast<const v8h*>(pa + 8);
        const _Float16* pb = Ws + (size_t)(n_blk + t) * GRP + kk;
        br0 = *reinterpret_cast<const v8h*>(pb);
        br1 = *reinterpret_cast<const v8h*>(pb + 8);
        br2 = *reinterpret_cast<const v8h*>(pb + 16);
        br3 = *reinterpret_cast<const v8h*>(pb + 24);
    };
    auto lstore = [&](int buf) {
        _Float16* pa = &As[buf][a_row * LDS_STRIDE + a_k];
        *reinterpret_cast<v8h*>(pa)     = ar0;
        *reinterpret_cast<v8h*>(pa + 8) = ar1;
        _Float16* pb = &Bs[buf][t * LDS_STRIDE];
        *reinterpret_cast<v8h*>(pb)      = br0;
        *reinterpret_cast<v8h*>(pb + 8)  = br1;
        *reinterpret_cast<v8h*>(pb + 16) = br2;
        *reinterpret_cast<v8h*>(pb + 24) = br3;
    };
#endif

    auto compute = [&](int buf) {
        v16h a[4], b[4];
        #pragma unroll
        for (int mi = 0; mi < 4; ++mi) {
            const _Float16* p =
                &As[buf][(wm * 64 + mi * 16 + lrow) * LDS_STRIDE + khalf];
            v8h lo = *reinterpret_cast<const v8h*>(p);       // K = khalf..+7
            v8h hi = *reinterpret_cast<const v8h*>(p + 16);  // K = khalf+16..+23
            #pragma unroll
            for (int q = 0; q < 8; ++q) { a[mi][q] = lo[q]; a[mi][q + 8] = hi[q]; }
        }
        #pragma unroll
        for (int ni = 0; ni < 4; ++ni) {
            const _Float16* p =
                &Bs[buf][(wn * 64 + ni * 16 + lrow) * LDS_STRIDE + khalf];
            v8h lo = *reinterpret_cast<const v8h*>(p);
            v8h hi = *reinterpret_cast<const v8h*>(p + 16);
            #pragma unroll
            for (int q = 0; q < 8; ++q) { b[ni][q] = lo[q]; b[ni][q + 8] = hi[q]; }
        }
        #pragma unroll
        for (int mi = 0; mi < 4; ++mi)
            #pragma unroll
            for (int ni = 0; ni < 4; ++ni)
                acc[mi][ni] = __builtin_amdgcn_wmma_f32_16x16x32_f16(
                    false, a[mi], false, b[ni],
                    (short)0, acc[mi][ni], false, false);
    };

    const int STEPS = GRP / BLK_K;           // 32

#if ASYNC_LDS
    stage(0, 0);
    __builtin_amdgcn_s_wait_asynccnt(0);
    __syncthreads();
    #pragma clang loop unroll(disable)
    for (int s = 0; s < STEPS; ++s) {
        if (s + 1 < STEPS) stage((s + 1) & 1, (s + 1) * BLK_K);
        compute(s & 1);
        if (s + 1 < STEPS) {
            __builtin_amdgcn_s_wait_asynccnt(0);
            __syncthreads();
        }
    }
#else
    gload(0);
    lstore(0);
    __syncthreads();
    #pragma clang loop unroll(disable)
    for (int s = 0; s < STEPS; ++s) {
        if (s + 1 < STEPS) gload((s + 1) * BLK_K);
        compute(s & 1);
        if (s + 1 < STEPS) {
            lstore((s + 1) & 1);
            __syncthreads();
        }
    }
#endif

    // Epilogue: scale + bias.  C layout: VGPR r -> M = r + 8*(lane>=16), N = lane%16
    const float sc = wscale[0] * (__uint_as_float(absmax_bits[0]) * (1.0f / 127.0f));
    const int rbase = (lane < 16) ? 0 : 8;
    const int m0 = m_blk + wm * 64;
    const int n0 = n_blk + wn * 64;
    #pragma unroll
    for (int mi = 0; mi < 4; ++mi) {
        #pragma unroll
        for (int ni = 0; ni < 4; ++ni) {
            const int col = n0 + ni * 16 + lrow;
            const float bv = bias[col];
            #pragma unroll
            for (int r = 0; r < 8; ++r) {
                const int row = m0 + mi * 16 + rbase + r;
                __builtin_nontemporal_store(acc[mi][ni][r] * sc + bv,
                                            &out[(size_t)row * OUT_F + col]);
            }
        }
    }
}

// ---------------------------------------------------------------------------
extern "C" void kernel_launch(void* const* d_in, const int* in_sizes, int n_in,
                              void* d_out, int out_size, void* d_ws, size_t ws_size,
                              hipStream_t stream) {
    const float* input  = (const float*)d_in[0];   // [8192, 4096] f32
    const float* weight = (const float*)d_in[1];   // [4096, 4096] f32 (ternary)
    const float* wscale = (const float*)d_in[2];   // [1] f32
    const float* bias   = (const float*)d_in[3];   // [4096] f32
    float* out = (float*)d_out;                    // [8192, 4096] f32

    unsigned int* absmax_bits = (unsigned int*)d_ws;
    _Float16* Xm = (_Float16*)((char*)d_ws + 256);
    _Float16* Ws = (_Float16*)((char*)d_ws + 256 + (size_t)N_ROWS * GRP * sizeof(_Float16));

    init_ws_kernel<<<1, 1, 0, stream>>>(absmax_bits);

    absmax_kernel<<<4096, 256, 0, stream>>>(input, absmax_bits, (N_ROWS * IN_F) / 4);

    const int xg = N_ROWS * GRP;   // 8,388,608 groups
    quant_group_kernel<<<xg / 256, 256, 0, stream>>>(input, absmax_bits, Xm, xg);

    const int wg = OUT_F * GRP;    // 4,194,304 groups
    wsum_group_kernel<<<wg / 256, 256, 0, stream>>>(weight, Ws, wg);

    dim3 grid(N_ROWS / BLK_M, OUT_F / BLK_N);  // 64 x 16 blocks
    bitlinear_wmma_gemm<<<grid, 256, 0, stream>>>(Xm, Ws, bias, wscale, absmax_bits, out);
}
